// CFConvTriple_37795712205372
// MI455X (gfx1250) — matrix operand order repaired and checked by reading, not compiled, not run on hardware
//
#include <hip/hip_runtime.h>

typedef __attribute__((ext_vector_type(16))) _Float16 v16h;
typedef __attribute__((ext_vector_type(8)))  float    v8f;
typedef float    f32x4 __attribute__((ext_vector_type(4)));
typedef f32x4    uf32x4 __attribute__((aligned(4)));   // dword-aligned 128-bit loads
typedef _Float16 h2v __attribute__((ext_vector_type(2)));

#define B_    2
#define A_    512
#define N_    1024
#define F_    64
#define DIN_  128
#define DOUT_ 128
#define TH_   25
#define CUT_  5.0f

// shifted softplus: ln(0.5*e^x + 0.5) = max(x,0) + ln2*log2(0.5 + 0.5*2^(-|x|*log2e))
// exp2 arg <= 0 -> result in (0,1]; log2 arg in (0.5,1] -> never subnormal/overflow,
// so raw v_exp_f32 / v_log_f32 are safe (no guard code).
__device__ __forceinline__ float sspf(float x) {
#if __has_builtin(__builtin_amdgcn_exp2f) && __has_builtin(__builtin_amdgcn_logf)
    float t = __builtin_amdgcn_exp2f(-1.44269504089f * __builtin_fabsf(x));
    float l = __builtin_amdgcn_logf(fmaf(0.5f, t, 0.5f));
    return fmaxf(x, 0.0f) + 0.69314718056f * l;
#else
    float ax = __builtin_fabsf(x);
    return fmaxf(x, 0.0f) + __logf(1.0f + __expf(-ax)) - 0.69314718056f;
#endif
}
__device__ __forceinline__ float ccut(float r) {
    float c = 0.5f * (__cosf(0.62831853071795865f * r) + 1.0f); // pi/CUTOFF
    return (r < CUT_) ? c : 0.0f;
}
__device__ __forceinline__ float frcp(float x) {
#if __has_builtin(__builtin_amdgcn_rcpf)
    return __builtin_amdgcn_rcpf(x);   // 1-ulp v_rcp_f32; args are O(1) positive
#else
    return 1.0f / x;
#endif
}

// ---------------------------------------------------------------------------
// Kernel 1: y = x @ W_in2f   (B*A=1024 rows, Din=128 -> F=64).  Tiny (17 MFLOP).
// ---------------------------------------------------------------------------
__global__ void cfconv_in2f_kernel(const float* __restrict__ x,
                                   const float* __restrict__ Wi,
                                   float* __restrict__ y) {
    unsigned idx = blockIdx.x * blockDim.x + threadIdx.x;   // over B*A*F
    unsigned row = idx >> 6;
    unsigned f   = idx & 63;
    const float* xr = x + row * DIN_;
    float acc = 0.f;
    #pragma unroll
    for (int d = 0; d < DIN_; ++d) acc = fmaf(xr[d], Wi[d * F_ + f], acc);
    y[idx] = acc;
}

// ---------------------------------------------------------------------------
// Kernel 2: fused triple-filter MLP (two WMMA GEMMs) + cutoffs + neighbor mix
//           + masked reduction over N + f2out.  One block per (b,a).
// ---------------------------------------------------------------------------
union V16 { v16h v; _Float16 e[16]; };
union V8F { v8f  v; float    e[8];  };

__launch_bounds__(256, 1)
__global__ void cfconv_triple_kernel(const float* __restrict__ r_ij,
                                     const float* __restrict__ r_ik,
                                     const int*   __restrict__ nbr_j,
                                     const int*   __restrict__ nbr_k,
                                     const float* __restrict__ tmask,
                                     const float* __restrict__ d_ijk,
                                     const float* __restrict__ W_t1,
                                     const float* __restrict__ b_t1,
                                     const float* __restrict__ W_t2,
                                     const float* __restrict__ b_t2,
                                     const float* __restrict__ W_f2out,
                                     const float* __restrict__ b_f2out,
                                     const float* __restrict__ y,
                                     float*       __restrict__ out) {
    // per-wave h staging tile, [f][m] layout so adjacent C-layout rows (j,j+1)
    // pack into one b32 store; m padded 16->18: write stride = 36B = 9 dword
    // banks, gcd(9,64)=1 -> conflict-free across the half-wave.
    __shared__ _Float16 hbuf[8][64][18];
    __shared__ float    accF[F_];

    const unsigned blk  = blockIdx.x;       // b*A + a
    const unsigned bIdx = blk >> 9;         // / A_
    const unsigned tid  = threadIdx.x;
    const unsigned wave = tid >> 5;
    const unsigned lane = tid & 31;
    const unsigned lo   = lane & 15;
    const unsigned hi   = lane >> 4;

    if (tid < F_) accF[tid] = 0.0f;
    __syncthreads();

    // ---- loop-invariant WMMA B fragments --------------------------------
    // B layout (f16 KxN=32x16): N = lane%16, half hh holds K = (hh<8?0:16) + hi*8 + (hh&7)
    V16 b1[4];          // W_t1 (25x64 padded to 32x64), 4 f-tiles
    V16 b2[4][2];       // W_t2 (64x64), 4 f-tiles x 2 K-halves
    float bt1v[4], bt2v[4];
    #pragma unroll
    for (int t = 0; t < 4; ++t) {
        const unsigned f = t * 16 + lo;
        bt1v[t] = b_t1[f];
        bt2v[t] = b_t2[f];
        #pragma unroll
        for (int hh = 0; hh < 16; ++hh) {
            const unsigned k = ((hh < 8) ? 0 : 16) + hi * 8 + (hh & 7);
            b1[t].e[hh] = (k < TH_) ? (_Float16)W_t1[k * F_ + f] : (_Float16)0.0f;
        }
        #pragma unroll
        for (int kh = 0; kh < 2; ++kh)
            #pragma unroll
            for (int hh = 0; hh < 16; ++hh) {
                const unsigned k = kh * 32 + ((hh < 8) ? 0 : 16) + hi * 8 + (hh & 7);
                b2[t][kh].e[hh] = (_Float16)W_t2[k * F_ + f];
            }
    }

    float acc[4] = {0.f, 0.f, 0.f, 0.f};   // per-lane partial sum over triples

    for (unsigned nb = 0; nb < N_; nb += 128) {
        const unsigned nBase = nb + wave * 16;
        const float* drow = d_ijk + blk * (unsigned)(N_ * TH_) + (nBase + lo) * TH_;

        // prefetch next tile of d_ijk while we compute on this one
        if (nb + 128 < N_)
            __builtin_prefetch(drow + 128 * TH_, 0, 1);

        // ---- A1 fragment: d_ijk rows nBase..nBase+15, K padded 25 -> 32 ----
        // A layout (f16 MxK=16x32): M = lane%16; lanes 0-15 hold K {0-7,16-23},
        // lanes 16-31 hold K {8-15, 24(+zeros)}.  Contiguous -> b128 loads.
        f32x4 q0, q1, q2, q3;
        if (hi == 0) {
            q0 = *(const uf32x4*)(drow + 0);
            q1 = *(const uf32x4*)(drow + 4);
            q2 = *(const uf32x4*)(drow + 16);
            q3 = *(const uf32x4*)(drow + 20);
        } else {
            q0 = *(const uf32x4*)(drow + 8);
            q1 = *(const uf32x4*)(drow + 12);
            q2 = (f32x4){drow[24], 0.f, 0.f, 0.f};
            q3 = (f32x4){0.f, 0.f, 0.f, 0.f};
        }
        V16 a1;
        #pragma unroll
        for (int i = 0; i < 4; ++i) {
            a1.e[i]      = (_Float16)q0[i];
            a1.e[4 + i]  = (_Float16)q1[i];
            a1.e[8 + i]  = (_Float16)q2[i];
            a1.e[12 + i] = (_Float16)q3[i];
        }

        // ---- GEMM1 + bias + ssp -> h staged in LDS (wave-private region) ---
        V8F g1[4];
        #pragma unroll
        for (int t = 0; t < 4; ++t) {
            v8f c = {};
            c = __builtin_amdgcn_wmma_f32_16x16x32_f16(false, a1.v, false, b1[t].v,
                                                       (short)0, c, false, false);
            g1[t].v = c;
        }
        // independent ssp chains across all 32 elements, then packed b32 stores
        _Float16 hv[4][8];
        #pragma unroll
        for (int t = 0; t < 4; ++t)
            #pragma unroll
            for (int j = 0; j < 8; ++j)
                hv[t][j] = (_Float16)sspf(g1[t].e[j] + bt1v[t]);
        #pragma unroll
        for (int t = 0; t < 4; ++t)
            #pragma unroll
            for (int j = 0; j < 8; j += 2) {           // rows m = j+8*hi, m+1
                h2v p; p[0] = hv[t][j]; p[1] = hv[t][j + 1];
                *(h2v*)&hbuf[wave][t * 16 + lo][j + 8 * hi] = p;
            }
        __builtin_amdgcn_wave_barrier();   // keep LDS store->load ordering (wave-local)

        // ---- transpose read: A2 fragments (M = lane%16 row of h, K = f) ----
        V16 a2[2];
        #pragma unroll
        for (int kh = 0; kh < 2; ++kh)
            #pragma unroll
            for (int hh = 0; hh < 16; ++hh) {
                const unsigned f = kh * 32 + ((hh < 8) ? 0 : 16) + hi * 8 + (hh & 7);
                a2[kh].e[hh] = hbuf[wave][f][lo];
            }

        // ---- GEMM2: W_triple tile (pre-bias) -------------------------------
        V8F cw[4];
        #pragma unroll
        for (int t = 0; t < 4; ++t) {
            v8f c = {};
            c = __builtin_amdgcn_wmma_f32_16x16x32_f16(false, a2[0].v, false, b2[t][0].v,
                                                       (short)0, c, false, false);
            c = __builtin_amdgcn_wmma_f32_16x16x32_f16(false, a2[1].v, false, b2[t][1].v,
                                                       (short)0, c, false, false);
            cw[t].v = c;
        }

        // ---- fused epilogue: cutoffs * neighbor mix * mask, reduce over n --
        #pragma unroll
        for (int j = 0; j < 8; ++j) {
            const unsigned m = j + 8 * hi;
            const unsigned g = blk * N_ + (nBase + m);
            const float rij = r_ij[g];
            const float rik = r_ik[g];
            const float msk = tmask[g];
            const int   nj  = nbr_j[g];
            const int   nk  = nbr_k[g];
            const float w   = msk * ccut(rij) * ccut(rik);
            const float inv = frcp(rij + rik);
            const float* yj = y + (bIdx * A_ + (unsigned)nj) * F_;
            const float* yk = y + (bIdx * A_ + (unsigned)nk) * F_;
            #pragma unroll
            for (int t = 0; t < 4; ++t) {
                const unsigned f = t * 16 + lo;
                const float mix = (rij * yj[f] + rik * yk[f]) * inv;
                acc[t] = fmaf(w * mix, cw[t].e[j] + bt2v[t], acc[t]);
            }
        }
    }

    // ---- reduce: lane pairs (l, l+16) share f; then across the 8 waves -----
    #pragma unroll
    for (int t = 0; t < 4; ++t) {
        float v = acc[t] + __shfl_xor(acc[t], 16, 32);
        if (hi == 0) atomicAdd(&accF[t * 16 + lo], v);   // ds_add_f32
    }
    __syncthreads();

    // ---- f2out + ssp: (1 x 64) @ (64 x 128) --------------------------------
    if (tid < DOUT_) {
        float s = b_f2out[tid];
        #pragma unroll
        for (int f = 0; f < F_; ++f) s = fmaf(accF[f], W_f2out[f * DOUT_ + tid], s);
        out[blk * DOUT_ + tid] = sspf(s);
    }
}

// ---------------------------------------------------------------------------
extern "C" void kernel_launch(void* const* d_in, const int* in_sizes, int n_in,
                              void* d_out, int out_size, void* d_ws, size_t ws_size,
                              hipStream_t stream) {
    const float* x       = (const float*)d_in[0];
    // d_in[1] r_double, d_in[4] r_jk, d_in[5] neighbors, d_in[6] neighbor_mask: unused
    const float* r_ij    = (const float*)d_in[2];
    const float* r_ik    = (const float*)d_in[3];
    const int*   nbr_j   = (const int*)  d_in[7];
    const int*   nbr_k   = (const int*)  d_in[8];
    const float* tmask   = (const float*)d_in[9];
    const float* d_ijk   = (const float*)d_in[10];
    const float* W_in2f  = (const float*)d_in[11];
    const float* W_t1    = (const float*)d_in[12];
    const float* b_t1    = (const float*)d_in[13];
    const float* W_t2    = (const float*)d_in[14];
    const float* b_t2    = (const float*)d_in[15];
    const float* W_f2out = (const float*)d_in[16];
    const float* b_f2out = (const float*)d_in[17];

    float* y   = (float*)d_ws;                 // B*A*F floats = 256 KB scratch
    float* out = (float*)d_out;                // (B, A, 128) f32

    (void)in_sizes; (void)n_in; (void)out_size; (void)ws_size;

    // Stage 1: y = x @ W_in2f
    cfconv_in2f_kernel<<<(B_ * A_ * F_) / 256, 256, 0, stream>>>(x, W_in2f, y);

    // Stage 2: fused triple pipeline, one block per (b,a)
    cfconv_triple_kernel<<<B_ * A_, 256, 0, stream>>>(
        r_ij, r_ik, nbr_j, nbr_k, tmask, d_ijk,
        W_t1, b_t1, W_t2, b_t2, W_f2out, b_f2out, y, out);
}
